// SimpleMLP_39762807226843
// MI455X (gfx1250) — matrix operand order repaired and checked by reading, not compiled, not run on hardware
//
#include <hip/hip_runtime.h>

typedef __attribute__((ext_vector_type(16))) _Float16 v16h;
typedef __attribute__((ext_vector_type(8)))  _Float16 v8h;
typedef __attribute__((ext_vector_type(8)))  float    v8f;

#define H 128
#define TILE_ATOMS 128
#define NLAYERS 2

__device__ __forceinline__ float silu_f(float x) {
    // x * sigmoid(x) with fast v_rcp_f32 instead of IEEE divide
    return x * __builtin_amdgcn_rcpf(1.0f + __expf(-x));
}

// ---------------------------------------------------------------------------
// Main fused MLP kernel: 128 atoms per workgroup, 8 waves, WMMA f16->f32.
// Writes per-atom energies to wsE[N] and per-WG h-column sums to wsHsum[wg*H].
// ---------------------------------------------------------------------------
__global__ __launch_bounds__(256) void mlp_main(
    const float* __restrict__ pos, const int* __restrict__ Zn,
    const float* __restrict__ Win, const float* __restrict__ bin,
    const float* __restrict__ Wh,  const float* __restrict__ bh,
    const float* __restrict__ We,  const float* __restrict__ bE,
    float* __restrict__ wsE, float* __restrict__ wsHsum, int N)
{
    __shared__ _Float16 sWT[NLAYERS * H * H];   // weights, transposed [l][n][k], 64 KB
    __shared__ _Float16 sH[TILE_ATOMS * H];     // activations, row-major [a][n], 32 KB
    __shared__ float sWin[4 * H];
    __shared__ float sBin[H];
    __shared__ float sBh[NLAYERS * H];
    __shared__ float sWe[H];

    const int t  = threadIdx.x;
    const int wg = blockIdx.x;
    const int a0 = wg * TILE_ATOMS;

    // ---- Stage + transpose weights (f32 -> f16) -------------------------
    for (int idx = t; idx < NLAYERS * H * H; idx += 256) {
        int l   = idx >> 14;           // H*H == 16384
        int rem = idx & (H * H - 1);
        int k   = rem >> 7;
        int n   = rem & (H - 1);
        sWT[l * H * H + n * H + k] = (_Float16)Wh[idx];
    }
    for (int idx = t; idx < 4 * H; idx += 256) sWin[idx] = Win[idx];
    for (int idx = t; idx < H; idx += 256) { sBin[idx] = bin[idx]; sWe[idx] = We[idx]; }
    for (int idx = t; idx < NLAYERS * H; idx += 256) sBh[idx] = bh[idx];
    __syncthreads();

    // ---- Layer 0: feats[128x4] @ W_in[4x128] + silu, into sH (f16) ------
    {
        int aLoc  = t >> 1;              // wave w covers its own 16 rows
        int jBase = (t & 1) * 64;
        int aG    = a0 + aLoc;
        float f0 = 0.f, f1 = 0.f, f2 = 0.f, f3 = 0.f;
        if (aG < N) {
            f0 = pos[aG * 3 + 0];
            f1 = pos[aG * 3 + 1];
            f2 = pos[aG * 3 + 2];
            f3 = (float)Zn[aG];
        }
        #pragma unroll
        for (int jv = 0; jv < 8; ++jv) {
            v8h pack;
            #pragma unroll
            for (int e = 0; e < 8; ++e) {
                int j = jBase + jv * 8 + e;
                float x = sBin[j] + f0 * sWin[j] + f1 * sWin[H + j]
                                  + f2 * sWin[2 * H + j] + f3 * sWin[3 * H + j];
                pack[e] = (_Float16)silu_f(x);
            }
            *(v8h*)(sH + aLoc * H + jBase + jv * 8) = pack;
        }
    }

    // ---- Hidden layers: h = silu(h @ W_h[l] + b_h[l]) via WMMA ----------
    const int lane = t & 31;
    const int wave = t >> 5;
    const int col  = lane & 15;
    const int hiHalf = lane >> 4;              // 0: lanes 0-15, 1: lanes 16-31
    const int rowA = wave * 16 + col;          // this lane's A-matrix row (atom)

    for (int l = 0; l < NLAYERS; ++l) {
        // Load full A row (K=0..127) for this lane, contiguous k order (matches B):
        // lanes 0-15 hold K = kc*32 + {0..15}; lanes 16-31 same +16.
        v16h Afrag[4];
        const _Float16* arow = sH + rowA * H + hiHalf * 16;
        #pragma unroll
        for (int kc = 0; kc < 4; ++kc) {
            v8h lo = *(const v8h*)(arow + kc * 32);
            v8h hi = *(const v8h*)(arow + kc * 32 + 8);
            Afrag[kc] = __builtin_shufflevector(lo, hi,
                0, 1, 2, 3, 4, 5, 6, 7, 8, 9, 10, 11, 12, 13, 14, 15);
        }
        // All A data of this wave is now in VGPRs -> safe to overwrite sH rows.

        const _Float16* bbase = sWT + l * H * H + hiHalf * 16;
        // Double-buffered B fragments: prefetch tile nt+1 while tile nt computes.
        v16h Bfrag[2][4];
        #pragma unroll
        for (int kc = 0; kc < 4; ++kc)
            Bfrag[0][kc] = *(const v16h*)(bbase + col * H + kc * 32);

        #pragma unroll
        for (int nt = 0; nt < 8; ++nt) {
            int n = nt * 16 + col;
            if (nt < 7) {
                int n2 = (nt + 1) * 16 + col;
                #pragma unroll
                for (int kc = 0; kc < 4; ++kc)
                    Bfrag[(nt + 1) & 1][kc] = *(const v16h*)(bbase + n2 * H + kc * 32);
            }
            float bias = sBh[l * H + n];
            v8f acc = {bias, bias, bias, bias, bias, bias, bias, bias};
            #pragma unroll
            for (int kc = 0; kc < 4; ++kc) {
                acc = __builtin_amdgcn_wmma_f32_16x16x32_f16(
                    false, Afrag[kc], false, Bfrag[nt & 1][kc], (short)0, acc,
                    false, false);
            }
            #pragma unroll
            for (int v = 0; v < 8; ++v) {
                int m = hiHalf * 8 + v;        // C/D layout: VGPR v -> row m
                sH[(wave * 16 + m) * H + n] = (_Float16)silu_f(acc[v]);
            }
        }
    }
    __syncthreads();

    // ---- Heads: per-atom energy and per-WG h column sums ----------------
    int nValid = N - a0;
    if (nValid > TILE_ATOMS) nValid = TILE_ATOMS;
    if (t < TILE_ATOMS) {
        if (t < nValid) {
            float e = bE[0];
            const _Float16* hrow = sH + t * H;
            #pragma unroll 8
            for (int n = 0; n < H; ++n) e += (float)hrow[n] * sWe[n];
            wsE[a0 + t] = e;
        }
        float s = 0.f;
        for (int a = 0; a < nValid; ++a) s += (float)sH[a * H + t];
        wsHsum[wg * H + t] = s;
    }
}

// ---------------------------------------------------------------------------
// Deterministic segment sum: batch is sorted, one thread per structure does a
// binary search for its [start,end) range and sums atom energies.
// ---------------------------------------------------------------------------
__global__ void seg_energy(const int* __restrict__ batch,
                           const float* __restrict__ wsE,
                           float* __restrict__ outE, int N, int S)
{
    int s = blockIdx.x * blockDim.x + threadIdx.x;
    if (s >= S) return;
    int lo = 0, hi = N;
    while (lo < hi) { int mid = (lo + hi) >> 1; if (batch[mid] < s) lo = mid + 1; else hi = mid; }
    int start = lo;
    hi = N;
    while (lo < hi) { int mid = (lo + hi) >> 1; if (batch[mid] <= s) lo = mid + 1; else hi = mid; }
    int end = lo;
    float sum = 0.f;
    for (int i = start; i < end; ++i) sum += wsE[i];
    outE[s] = sum;
}

__global__ void zero_forces(float* __restrict__ f, int n)
{
    int i = blockIdx.x * blockDim.x + threadIdx.x;
    if (i < n) f[i] = 0.f;
}

// ---------------------------------------------------------------------------
// Stress: reduce per-WG h sums -> mean over atoms -> @ W_s + b_s (6 outputs).
// ---------------------------------------------------------------------------
__global__ __launch_bounds__(128) void stress_kernel(
    const float* __restrict__ wsHsum, int nwg,
    const float* __restrict__ Ws, const float* __restrict__ bs,
    float* __restrict__ outS, int N)
{
    __shared__ float hmean[H];
    int t = threadIdx.x;
    float s = 0.f;
    for (int w = 0; w < nwg; ++w) s += wsHsum[w * H + t];
    hmean[t] = s / (float)N;
    __syncthreads();
    if (t < 6) {
        float acc = bs[t];
        #pragma unroll 8
        for (int n = 0; n < H; ++n) acc += hmean[n] * Ws[n * 6 + t];
        outS[t] = acc;
    }
}

extern "C" void kernel_launch(void* const* d_in, const int* in_sizes, int n_in,
                              void* d_out, int out_size, void* d_ws, size_t ws_size,
                              hipStream_t stream)
{
    const float* pos   = (const float*)d_in[0];
    const int*   Zn    = (const int*)  d_in[1];
    const int*   batch = (const int*)  d_in[2];
    const float* Win   = (const float*)d_in[3];
    const float* bin   = (const float*)d_in[4];
    const float* Wh    = (const float*)d_in[5];
    const float* bh    = (const float*)d_in[6];
    const float* We    = (const float*)d_in[7];
    const float* bE    = (const float*)d_in[8];
    const float* Ws    = (const float*)d_in[9];
    const float* bs    = (const float*)d_in[10];

    const int N   = in_sizes[1];                 // atoms
    const int S   = out_size - 3 * N - 6;        // structures
    const int nwg = (N + TILE_ATOMS - 1) / TILE_ATOMS;

    float* outE = (float*)d_out;                 // [S]
    float* outF = outE + S;                      // [N*3] zeros
    float* outS = outE + S + 3 * N;              // [6]

    float* wsE    = (float*)d_ws;                // [N] atom energies
    float* wsHsum = wsE + N;                     // [nwg*H] partial h sums

    mlp_main<<<nwg, 256, 0, stream>>>(pos, Zn, Win, bin, Wh, bh, We, bE,
                                      wsE, wsHsum, N);
    seg_energy<<<(S + 127) / 128, 128, 0, stream>>>(batch, wsE, outE, N, S);
    zero_forces<<<(3 * N + 255) / 256, 256, 0, stream>>>(outF, 3 * N);
    stress_kernel<<<1, 128, 0, stream>>>(wsHsum, nwg, Ws, bs, outS, N);
}
